// TransformerModel_72138270703780
// MI455X (gfx1250) — matrix operand order repaired and checked by reading, not compiled, not run on hardware
//
#include <hip/hip_runtime.h>

typedef __attribute__((ext_vector_type(16))) _Float16 v16h;
typedef __attribute__((ext_vector_type(8)))  float    v8f;

// ---------------- model constants ----------------
enum : int {
  BSZ = 4, SEQ = 512, TGT = 256, DM = 512, NH = 8, HD = 64, FF = 2048, NL = 6
};

// ---------------- GEMM tile config ----------------
// All GEMMs in this model are tile-aligned: M%128==0, N%64==0, K%32==0.
#define BM 128
#define BN 64
#define BK 32
#define LDA_S 36   // halves per A row in LDS (32 + 4 pad; row = 72B, 8B aligned)
#define LDB_S 68   // halves per B row in LDS (64 + 4 pad; row = 136B, 8B aligned)

__device__ __forceinline__ v8f wmma16(v16h a, v16h b, v8f c) {
  // D = A(16x32 f16) * B(32x16 f16) + C(16x16 f32)
  return __builtin_amdgcn_wmma_f32_16x16x32_f16(
      /*neg_a=*/false, a, /*neg_b=*/false, b,
      /*c_mod=*/(short)0, c, /*reuse_a=*/false, /*reuse_b=*/false);
}

// A fragment: 16x32 (MxK).  lanes 0-15: M=lane, K pairs {0..7,16..23};
// lanes 16-31: M=lane-16, K pairs {8..15,24..31}.  (ISA 7.12.2)
__device__ __forceinline__ v16h load_a_frag(const _Float16* AsH, int mbase, int lane) {
  int m  = mbase + (lane & 15);
  int kb = (lane & 16) ? 8 : 0;
  const _Float16* rowp = AsH + m * LDA_S;
  v16h a;
#pragma unroll
  for (int i = 0; i < 8; ++i) {
    int k = ((i & 4) ? 16 : 0) + kb + 2 * (i & 3);
    a[2 * i]     = rowp[k];
    a[2 * i + 1] = rowp[k + 1];
  }
  return a;
}

// B fragment: 32x16 (KxN), LDS stored K-major: N = lane&15, K same order as A fragment.
__device__ __forceinline__ v16h load_b_frag(const _Float16* BsH, int nbase, int lane) {
  const _Float16* colp = BsH + nbase + (lane & 15);
  int kb = (lane & 16) ? 8 : 0;
  v16h b;
#pragma unroll
  for (int i = 0; i < 16; ++i) b[i] = colp[(((i & 8) ? 16 : 0) + kb + (i & 7)) * LDB_S];
  return b;
}

// Batched GEMM:  C[M,N] = act( (A[M,K] . W^T + bias) * alpha + resid )
//   W stored [N,K] row-major when bTrans==0 (weights / Q.K^T with K-matrix rows),
//   W stored [K,N] row-major when bTrans==1 (attention . V).
__global__ __launch_bounds__(256)
void gemm_wmma_kernel(const float* __restrict__ A, int lda, long long sA,
                      const float* __restrict__ Bm, int ldb, long long sB, int bTrans,
                      float* __restrict__ C, int ldc, long long sC,
                      const float* __restrict__ bias,
                      const float* __restrict__ resid,
                      int M, int N, int K, float alpha, int doRelu) {
  __shared__ __align__(16) _Float16 AsH[BM * LDA_S];
  __shared__ __align__(16) _Float16 BsH[BK * LDB_S];

  const int bz = blockIdx.z;
  A  += (size_t)bz * sA;
  Bm += (size_t)bz * sB;
  C  += (size_t)bz * sC;

  const int m0 = blockIdx.y * BM;
  const int n0 = blockIdx.x * BN;
  const int tid  = threadIdx.x;
  const int lane = tid & 31;
  const int wave = tid >> 5;            // 8 waves
  const int wm = (wave >> 1) * 32;      // 4 waves along M
  const int wn = (wave & 1) * 32;       // 2 waves along N

  v8f acc00 = {}, acc01 = {}, acc10 = {}, acc11 = {};

  // per-thread staging registers (b128 global loads)
  float4 aR[4];
  float4 bR[2];

  auto load_tiles = [&](int kb) {
#pragma unroll
    for (int r = 0; r < 4; ++r) {
      int i  = tid + 256 * r;       // 0..1023
      int m  = i >> 3;              // 0..127
      int c4 = (i & 7) * 4;         // 0,4,..28
      aR[r] = *(const float4*)(A + (size_t)(m0 + m) * lda + (kb + c4));
    }
    if (!bTrans) {
#pragma unroll
      for (int r = 0; r < 2; ++r) {
        int i  = tid + 256 * r;     // 0..511
        int n  = i >> 3;            // 0..63
        int c4 = (i & 7) * 4;
        bR[r] = *(const float4*)(Bm + (size_t)(n0 + n) * ldb + (kb + c4));
      }
    } else {
#pragma unroll
      for (int r = 0; r < 2; ++r) {
        int i  = tid + 256 * r;
        int k  = i >> 4;            // 0..31
        int n4 = (i & 15) * 4;      // 0..60
        bR[r] = *(const float4*)(Bm + (size_t)(kb + k) * ldb + (n0 + n4));
      }
    }
  };

  auto store_tiles = [&]() {
#pragma unroll
    for (int r = 0; r < 4; ++r) {
      int i  = tid + 256 * r;
      int m  = i >> 3;
      int c4 = (i & 7) * 4;
      union { _Float16 h[4]; uint2 u; } t;
      t.h[0] = (_Float16)aR[r].x; t.h[1] = (_Float16)aR[r].y;
      t.h[2] = (_Float16)aR[r].z; t.h[3] = (_Float16)aR[r].w;
      *(uint2*)&AsH[m * LDA_S + c4] = t.u;   // ds_store_b64, 8B aligned
    }
    if (!bTrans) {
      // transpose N-major global -> K-major LDS
#pragma unroll
      for (int r = 0; r < 2; ++r) {
        int i  = tid + 256 * r;
        int n  = i >> 3;
        int c4 = (i & 7) * 4;
        BsH[(c4 + 0) * LDB_S + n] = (_Float16)bR[r].x;
        BsH[(c4 + 1) * LDB_S + n] = (_Float16)bR[r].y;
        BsH[(c4 + 2) * LDB_S + n] = (_Float16)bR[r].z;
        BsH[(c4 + 3) * LDB_S + n] = (_Float16)bR[r].w;
      }
    } else {
      // already K-major: packed 64-bit stores
#pragma unroll
      for (int r = 0; r < 2; ++r) {
        int i  = tid + 256 * r;
        int k  = i >> 4;
        int n4 = (i & 15) * 4;
        union { _Float16 h[4]; uint2 u; } t;
        t.h[0] = (_Float16)bR[r].x; t.h[1] = (_Float16)bR[r].y;
        t.h[2] = (_Float16)bR[r].z; t.h[3] = (_Float16)bR[r].w;
        *(uint2*)&BsH[k * LDB_S + n4] = t.u;
      }
    }
  };

  load_tiles(0);
  for (int kb = 0; kb < K; kb += BK) {
    __syncthreads();           // previous compute finished reading LDS
    store_tiles();
    __syncthreads();
    if (kb + BK < K) {
      load_tiles(kb + BK);     // in flight while WMMAs run below
      if (kb + 2 * BK < K) {   // L2 prefetch two tiles ahead (global_prefetch_b8)
        __builtin_prefetch(A + (size_t)(m0 + (tid & 127)) * lda + (kb + 2 * BK), 0, 1);
      }
    }

    v16h a0 = load_a_frag(AsH, wm + 0,  lane);
    v16h a1 = load_a_frag(AsH, wm + 16, lane);
    v16h b0 = load_b_frag(BsH, wn + 0,  lane);
    v16h b1 = load_b_frag(BsH, wn + 16, lane);

    acc00 = wmma16(a0, b0, acc00);
    acc01 = wmma16(a0, b1, acc01);
    acc10 = wmma16(a1, b0, acc10);
    acc11 = wmma16(a1, b1, acc11);
  }

  // ---- epilogue: C/D layout -> VGPR v: M = v + 8*(lane>=16), N = lane&15 ----
  const int mofs = (lane >> 4) * 8;
  const int ncol = lane & 15;
#pragma unroll
  for (int t = 0; t < 4; ++t) {
    v8f acc = (t == 0) ? acc00 : (t == 1) ? acc01 : (t == 2) ? acc10 : acc11;
    int msub = (t >> 1) * 16;
    int nsub = (t & 1) * 16;
    int col = n0 + wn + nsub + ncol;
    float bv = bias ? bias[col] : 0.0f;
#pragma unroll
    for (int v = 0; v < 8; ++v) {
      int row = m0 + wm + msub + mofs + v;
      float val = (acc[v] + bv) * alpha;
      if (resid) val += resid[(size_t)row * ldc + col];
      if (doRelu) val = fmaxf(val, 0.0f);
      C[(size_t)row * ldc + col] = val;
    }
  }
}

// x[b,s,:] += pe[pos], pos = usePos ? s : 0   (d = 512)
__global__ void add_pe_kernel(float* __restrict__ X, int total, int Slen, int usePos) {
  int idx = blockIdx.x * 256 + threadIdx.x;
  if (idx >= total) return;
  int i = idx & (DM - 1);
  int s = (idx >> 9) % Slen;
  float pos = usePos ? (float)s : 0.0f;
  float dv  = expf(-(float)(i & ~1) * (9.210340371976184f / (float)DM)); // ln(1e4)/d
  float ang = pos * dv;
  X[idx] += (i & 1) ? cosf(ang) : sinf(ang);
}

// src[M, C] column slice [off, off+512) -> dst[(b*NH+h)*Slen + s, 64]
__global__ void split_heads_kernel(const float* __restrict__ src, int C, int off,
                                   float* __restrict__ dst, int Slen, int total) {
  int idx = blockIdx.x * 256 + threadIdx.x;
  if (idx >= total) return;
  int e = idx & 63;
  int rest = idx >> 6;
  int s = rest % Slen;
  int bh = rest / Slen;
  int h = bh & 7;
  int b = bh >> 3;
  dst[idx] = src[(size_t)(b * Slen + s) * C + off + h * HD + e];
}

// dst[b*Slen+s, 512] <- src[(b*NH+h)*Slen+s, 64]
__global__ void merge_heads_kernel(const float* __restrict__ src,
                                   float* __restrict__ dst, int Slen, int total) {
  int idx = blockIdx.x * 256 + threadIdx.x;
  if (idx >= total) return;
  int e = idx & 63;
  int h = (idx >> 6) & 7;
  int rest = idx >> 9;
  int s = rest % Slen;
  int b = rest / Slen;
  dst[idx] = src[((size_t)(b * NH + h) * Slen + s) * HD + e];
}

// row softmax with optional causal mask (q = row % Sq, mask cols > q)
__global__ __launch_bounds__(256)
void softmax_kernel(float* __restrict__ Sm, int Sq, int Sk, int causal) {
  int row = blockIdx.x;
  float* p = Sm + (size_t)row * Sk;
  int q = row % Sq;
  __shared__ float red[256];
  int t = threadIdx.x;

  float lmax = -1e30f;
  for (int c = t; c < Sk; c += 256) {
    float v = p[c];
    if (causal && c > q) v = -1e9f;
    lmax = fmaxf(lmax, v);
  }
  red[t] = lmax; __syncthreads();
  for (int s = 128; s > 0; s >>= 1) {
    if (t < s) red[t] = fmaxf(red[t], red[t + s]);
    __syncthreads();
  }
  float m = red[0]; __syncthreads();

  float lsum = 0.0f;
  for (int c = t; c < Sk; c += 256) {
    float v = p[c];
    if (causal && c > q) v = -1e9f;
    float e = expf(v - m);
    p[c] = e;
    lsum += e;
  }
  red[t] = lsum; __syncthreads();
  for (int s = 128; s > 0; s >>= 1) {
    if (t < s) red[t] += red[t + s];
    __syncthreads();
  }
  float inv = 1.0f / red[0];
  for (int c = t; c < Sk; c += 256) p[c] *= inv;
}

// in-place LayerNorm over rows of 512
__global__ __launch_bounds__(256)
void ln_kernel(float* __restrict__ X, const float* __restrict__ g,
               const float* __restrict__ b) {
  int row = blockIdx.x;
  float* p = X + (size_t)row * DM;
  int t = threadIdx.x;
  float x0 = p[t], x1 = p[t + 256];
  __shared__ float red[256];
  red[t] = x0 + x1; __syncthreads();
  for (int s = 128; s > 0; s >>= 1) { if (t < s) red[t] += red[t + s]; __syncthreads(); }
  float mean = red[0] * (1.0f / (float)DM); __syncthreads();
  float d0 = x0 - mean, d1 = x1 - mean;
  red[t] = d0 * d0 + d1 * d1; __syncthreads();
  for (int s = 128; s > 0; s >>= 1) { if (t < s) red[t] += red[t + s]; __syncthreads(); }
  float inv = rsqrtf(red[0] * (1.0f / (float)DM) + 1e-5f);
  p[t]       = d0 * inv * g[t] + b[t];
  p[t + 256] = d1 * inv * g[t + 256] + b[t + 256];
}

// ---------------------------------------------------------------------------
extern "C" void kernel_launch(void* const* d_in, const int* in_sizes, int n_in,
                              void* d_out, int out_size, void* d_ws, size_t ws_size,
                              hipStream_t stream) {
  (void)in_sizes; (void)n_in; (void)out_size; (void)ws_size;
  const float* src       = (const float*)d_in[0];
  const float* tgt       = (const float*)d_in[1];
  const float* enc_in_w  = (const float*)d_in[2];
  const float* enc_in_b  = (const float*)d_in[3];
  const float* dec_in_w  = (const float*)d_in[4];
  const float* dec_in_b  = (const float*)d_in[5];
  const float* out_w     = (const float*)d_in[6];
  const float* out_b     = (const float*)d_in[7];
  const float* enc_attn_w = (const float*)d_in[8];
  const float* enc_attn_b = (const float*)d_in[9];
  const float* enc_out_w  = (const float*)d_in[10];
  const float* enc_out_b  = (const float*)d_in[11];
  const float* enc_ff1_w  = (const float*)d_in[12];
  const float* enc_ff1_b  = (const float*)d_in[13];
  const float* enc_ff2_w  = (const float*)d_in[14];
  const float* enc_ff2_b  = (const float*)d_in[15];
  const float* enc_ln1_g  = (const float*)d_in[16];
  const float* enc_ln1_b  = (const float*)d_in[17];
  const float* enc_ln2_g  = (const float*)d_in[18];
  const float* enc_ln2_b  = (const float*)d_in[19];
  const float* dec_sa_w     = (const float*)d_in[20];
  const float* dec_sa_b     = (const float*)d_in[21];
  const float* dec_sa_out_w = (const float*)d_in[22];
  const float* dec_sa_out_b = (const float*)d_in[23];
  const float* dec_ca_w     = (const float*)d_in[24];
  const float* dec_ca_b     = (const float*)d_in[25];
  const float* dec_ca_out_w = (const float*)d_in[26];
  const float* dec_ca_out_b = (const float*)d_in[27];
  const float* dec_ff1_w    = (const float*)d_in[28];
  const float* dec_ff1_b    = (const float*)d_in[29];
  const float* dec_ff2_w    = (const float*)d_in[30];
  const float* dec_ff2_b    = (const float*)d_in[31];
  const float* dec_ln1_g    = (const float*)d_in[32];
  const float* dec_ln1_b    = (const float*)d_in[33];
  const float* dec_ln2_g    = (const float*)d_in[34];
  const float* dec_ln2_b    = (const float*)d_in[35];
  const float* dec_ln3_g    = (const float*)d_in[36];
  const float* dec_ln3_b    = (const float*)d_in[37];

  // ---- workspace carve-up (floats) ----
  float* ws = (float*)d_ws;
  float* XE = ws; ws += (size_t)BSZ * SEQ * DM;        // encoder stream / memory
  float* Y  = ws; ws += (size_t)BSZ * TGT * DM;        // decoder stream
  float* T1 = ws; ws += (size_t)BSZ * SEQ * FF;        // big temp (qkv / ff hidden / kv)
  float* T2 = ws; ws += (size_t)BSZ * TGT * DM;        // cross-attn q temp
  float* Qh = ws; ws += (size_t)BSZ * NH * SEQ * HD;
  float* Kh = ws; ws += (size_t)BSZ * NH * SEQ * HD;
  float* Vh = ws; ws += (size_t)BSZ * NH * SEQ * HD;
  float* AO = ws; ws += (size_t)BSZ * NH * SEQ * HD;
  float* SC = ws; ws += (size_t)BSZ * NH * SEQ * SEQ;
  float* MG = ws; ws += (size_t)BSZ * SEQ * DM;

  const float SQD = 22.627416997969522f;  // sqrt(512)
  const float ISQH = 0.125f;              // 1/sqrt(64)

  auto gemm = [&](const float* A, int lda, long long sA,
                  const float* Bm, int ldb, long long sB, int bTrans,
                  float* C, int ldc, long long sC,
                  const float* bias, const float* resid,
                  int M, int N, int K, int batch, float alpha, int relu) {
    dim3 g((N + BN - 1) / BN, (M + BM - 1) / BM, batch);
    gemm_wmma_kernel<<<g, dim3(256), 0, stream>>>(
        A, lda, sA, Bm, ldb, sB, bTrans, C, ldc, sC, bias, resid,
        M, N, K, alpha, relu);
  };
  auto split = [&](const float* s_, int C, int off, float* d_, int Slen) {
    int total = BSZ * NH * Slen * HD;
    split_heads_kernel<<<(total + 255) / 256, 256, 0, stream>>>(s_, C, off, d_, Slen, total);
  };
  auto merge = [&](const float* s_, float* d_, int Slen) {
    int total = BSZ * Slen * DM;
    merge_heads_kernel<<<(total + 255) / 256, 256, 0, stream>>>(s_, d_, Slen, total);
  };
  auto softmax = [&](float* s_, int Sq, int Sk, int causal) {
    softmax_kernel<<<BSZ * NH * Sq, 256, 0, stream>>>(s_, Sq, Sk, causal);
  };
  auto lnorm = [&](float* x, const float* g_, const float* b_, int rows) {
    ln_kernel<<<rows, 256, 0, stream>>>(x, g_, b_);
  };

  // ================= encoder =================
  const int ME = BSZ * SEQ;  // 2048
  gemm(src, HD, 0, enc_in_w, HD, 0, 0, XE, DM, 0, enc_in_b, nullptr,
       ME, DM, HD, 1, SQD, 0);
  add_pe_kernel<<<(ME * DM + 255) / 256, 256, 0, stream>>>(XE, ME * DM, SEQ, 1);

  for (int l = 0; l < NL; ++l) {
    const float* Wqkv = enc_attn_w + (size_t)l * 3 * DM * DM;
    const float* bqkv = enc_attn_b + (size_t)l * 3 * DM;
    // qkv projection
    gemm(XE, DM, 0, Wqkv, DM, 0, 0, T1, 3 * DM, 0, bqkv, nullptr,
         ME, 3 * DM, DM, 1, 1.0f, 0);
    split(T1, 3 * DM, 0,      Qh, SEQ);
    split(T1, 3 * DM, DM,     Kh, SEQ);
    split(T1, 3 * DM, 2 * DM, Vh, SEQ);
    // scores = Q K^T / 8   (batched over 32 heads)
    gemm(Qh, HD, (long long)SEQ * HD, Kh, HD, (long long)SEQ * HD, 0,
         SC, SEQ, (long long)SEQ * SEQ, nullptr, nullptr,
         SEQ, SEQ, HD, BSZ * NH, ISQH, 0);
    softmax(SC, SEQ, SEQ, 0);
    // attn . V
    gemm(SC, SEQ, (long long)SEQ * SEQ, Vh, HD, (long long)SEQ * HD, 1,
         AO, HD, (long long)SEQ * HD, nullptr, nullptr,
         SEQ, HD, SEQ, BSZ * NH, 1.0f, 0);
    merge(AO, MG, SEQ);
    // out projection + residual
    gemm(MG, DM, 0, enc_out_w + (size_t)l * DM * DM, DM, 0, 0,
         XE, DM, 0, enc_out_b + (size_t)l * DM, XE, ME, DM, DM, 1, 1.0f, 0);
    lnorm(XE, enc_ln1_g + (size_t)l * DM, enc_ln1_b + (size_t)l * DM, ME);
    // FFN
    gemm(XE, DM, 0, enc_ff1_w + (size_t)l * FF * DM, DM, 0, 0,
         T1, FF, 0, enc_ff1_b + (size_t)l * FF, nullptr, ME, FF, DM, 1, 1.0f, 1);
    gemm(T1, FF, 0, enc_ff2_w + (size_t)l * DM * FF, FF, 0, 0,
         XE, DM, 0, enc_ff2_b + (size_t)l * DM, XE, ME, DM, FF, 1, 1.0f, 0);
    lnorm(XE, enc_ln2_g + (size_t)l * DM, enc_ln2_b + (size_t)l * DM, ME);
  }
  // XE is now the encoder memory.

  // ================= decoder =================
  const int MD = BSZ * TGT;  // 1024
  gemm(tgt, HD, 0, dec_in_w, HD, 0, 0, Y, DM, 0, dec_in_b, nullptr,
       MD, DM, HD, 1, SQD, 0);
  add_pe_kernel<<<(MD * DM + 255) / 256, 256, 0, stream>>>(Y, MD * DM, TGT, 0);

  for (int l = 0; l < NL; ++l) {
    // ---- masked self-attention ----
    gemm(Y, DM, 0, dec_sa_w + (size_t)l * 3 * DM * DM, DM, 0, 0,
         T1, 3 * DM, 0, dec_sa_b + (size_t)l * 3 * DM, nullptr,
         MD, 3 * DM, DM, 1, 1.0f, 0);
    split(T1, 3 * DM, 0,      Qh, TGT);
    split(T1, 3 * DM, DM,     Kh, TGT);
    split(T1, 3 * DM, 2 * DM, Vh, TGT);
    gemm(Qh, HD, (long long)TGT * HD, Kh, HD, (long long)TGT * HD, 0,
         SC, TGT, (long long)TGT * TGT, nullptr, nullptr,
         TGT, TGT, HD, BSZ * NH, ISQH, 0);
    softmax(SC, TGT, TGT, 1);
    gemm(SC, TGT, (long long)TGT * TGT, Vh, HD, (long long)TGT * HD, 1,
         AO, HD, (long long)TGT * HD, nullptr, nullptr,
         TGT, HD, TGT, BSZ * NH, 1.0f, 0);
    merge(AO, MG, TGT);
    gemm(MG, DM, 0, dec_sa_out_w + (size_t)l * DM * DM, DM, 0, 0,
         Y, DM, 0, dec_sa_out_b + (size_t)l * DM, Y, MD, DM, DM, 1, 1.0f, 0);
    lnorm(Y, dec_ln1_g + (size_t)l * DM, dec_ln1_b + (size_t)l * DM, MD);

    // ---- cross-attention ----
    const float* Wca = dec_ca_w + (size_t)l * 3 * DM * DM;
    const float* bca = dec_ca_b + (size_t)l * 3 * DM;
    gemm(Y, DM, 0, Wca, DM, 0, 0, T2, DM, 0, bca, nullptr,
         MD, DM, DM, 1, 1.0f, 0);                               // q = Y Wq^T
    gemm(XE, DM, 0, Wca + (size_t)DM * DM, DM, 0, 0,
         T1, 2 * DM, 0, bca + DM, nullptr, ME, 2 * DM, DM, 1, 1.0f, 0); // kv = mem Wkv^T
    split(T2, DM, 0,      Qh, TGT);
    split(T1, 2 * DM, 0,  Kh, SEQ);
    split(T1, 2 * DM, DM, Vh, SEQ);
    gemm(Qh, HD, (long long)TGT * HD, Kh, HD, (long long)SEQ * HD, 0,
         SC, SEQ, (long long)TGT * SEQ, nullptr, nullptr,
         TGT, SEQ, HD, BSZ * NH, ISQH, 0);
    softmax(SC, TGT, SEQ, 0);
    gemm(SC, SEQ, (long long)TGT * SEQ, Vh, HD, (long long)SEQ * HD, 1,
         AO, HD, (long long)TGT * HD, nullptr, nullptr,
         TGT, HD, SEQ, BSZ * NH, 1.0f, 0);
    merge(AO, MG, TGT);
    gemm(MG, DM, 0, dec_ca_out_w + (size_t)l * DM * DM, DM, 0, 0,
         Y, DM, 0, dec_ca_out_b + (size_t)l * DM, Y, MD, DM, DM, 1, 1.0f, 0);
    lnorm(Y, dec_ln2_g + (size_t)l * DM, dec_ln2_b + (size_t)l * DM, MD);

    // ---- FFN ----
    gemm(Y, DM, 0, dec_ff1_w + (size_t)l * FF * DM, DM, 0, 0,
         T1, FF, 0, dec_ff1_b + (size_t)l * FF, nullptr, MD, FF, DM, 1, 1.0f, 1);
    gemm(T1, FF, 0, dec_ff2_w + (size_t)l * DM * FF, FF, 0, 0,
         Y, DM, 0, dec_ff2_b + (size_t)l * DM, Y, MD, DM, FF, 1, 1.0f, 0);
    lnorm(Y, dec_ln3_g + (size_t)l * DM, dec_ln3_b + (size_t)l * DM, MD);
  }

  // ================= output projection =================
  gemm(Y, DM, 0, out_w, DM, 0, 0, (float*)d_out, HD, 0, out_b, nullptr,
       MD, HD, DM, 1, 1.0f, 0);
}
